// LPALM_1941325218460
// MI455X (gfx1250) — compile-verified, matrix operand order checked by reading, not run on hardware
//
#include <hip/hip_runtime.h>
#include <hip/hip_bf16.h>

// Problem constants (from reference): b=16, m=k=256, n=2048, T=10
#define B_   16
#define M_   256
#define K_   256
#define N_   2048
#define T_   10
#define NT   128        // n-tile width per workgroup (8 waves x 16 cols)
#define SP   264        // padded LDS row stride for n-major tiles (elements)
#define ST   136        // padded LDS row stride for m/k-major transposed tiles
#define WGT  256        // threads per workgroup (8 wave32)

typedef __attribute__((ext_vector_type(16))) __bf16 v16bf;
typedef __attribute__((ext_vector_type(8)))  __bf16 v8bf;
typedef __attribute__((ext_vector_type(8)))  float  v8f;

// Fragment whose K dimension is contiguous in memory at p:
// elements [0..7] at p, [8..15] at p+16 (ISA 16-bit A/B layout, khi folded into p).
__device__ __forceinline__ v16bf frag_row(const __bf16* p) {
  v8bf lo = *(const v8bf*)(p);
  v8bf hi = *(const v8bf*)(p + 16);
  return __builtin_shufflevector(lo, hi, 0,1,2,3,4,5,6,7,8,9,10,11,12,13,14,15);
}

__device__ __forceinline__ v8f wmma_bf16(v16bf a, v16bf b, v8f c) {
  // D = A(16x32 bf16) * B(32x16 bf16) + C(16x16 f32)
  return __builtin_amdgcn_wmma_f32_16x16x32_bf16(false, a, false, b, (short)0, c,
                                                 false, false);
}

// ---------------------------------------------------------------------------
// K0: init S=0, A=1/sqrt(m), Abf=bf16(A), grad=0, WbfT[t][k][m] = bf16(W[t][m][k])
// ---------------------------------------------------------------------------
__global__ void lpalm_init(float* __restrict__ S, float* __restrict__ A,
                           __bf16* __restrict__ Abf, float* __restrict__ gradA,
                           __bf16* __restrict__ WbfT, const float* __restrict__ W)
{
  size_t i      = (size_t)blockIdx.x * blockDim.x + threadIdx.x;
  size_t stride = (size_t)gridDim.x * blockDim.x;
  const float a0 = 0.0625f; // 1/sqrt(256)
  for (size_t p = i; p < (size_t)B_ * K_ * N_; p += stride) S[p] = 0.0f;
  for (size_t p = i; p < (size_t)B_ * M_ * K_; p += stride) {
    A[p] = a0; Abf[p] = (__bf16)a0; gradA[p] = 0.0f;
  }
  for (size_t p = i; p < (size_t)T_ * M_ * K_; p += stride) {
    size_t t  = p >> 16;             // 65536 = M_*K_
    int    kk = (int)((p >> 8) & 255);
    int    mm = (int)(p & 255);
    WbfT[p] = (__bf16)W[(t * M_ + mm) * K_ + kk];
  }
}

// ---------------------------------------------------------------------------
// K1: fused iteration body for one (b, n-tile):
//   R = X - A*S ; S = soft(S + W^T R) ; E = A*S - X ; grad += E * S^T (atomics)
// ---------------------------------------------------------------------------
__global__ __launch_bounds__(WGT, 1) void lpalm_iter(
    const float* __restrict__ X, const __bf16* __restrict__ WbfT,
    const float* __restrict__ theta, float* __restrict__ S,
    const __bf16* __restrict__ Abf, float* __restrict__ gradA, int t)
{
  // n-major tiles (K of WMMA frags runs along m/k): used by phases 1-3
  __shared__ __align__(16) __bf16 ldsS[NT * SP];   // S[n][k]    66 KB
  __shared__ __align__(16) __bf16 ldsR[NT * SP];   // R[n][m]    66 KB
  // transposed tiles (K of WMMA frags runs along n): used by phase 4
  __shared__ __align__(16) __bf16 ldsSt[K_ * ST];  // S[k][n]    68 KB
  __shared__ __align__(16) __bf16 ldsEt[M_ * ST];  // E[m][n]    68 KB

  const int tid  = threadIdx.x;
  const int lane = tid & 31;
  const int w    = tid >> 5;        // wave id 0..7
  const int nl   = lane & 15;
  const int hi8  = (lane >> 4) * 8; // K-offset (A/B frags) == M-offset (C/D tiles)
  const int nt   = blockIdx.x;      // n-tile 0..15
  const int b    = blockIdx.y;      // batch 0..15
  const int cw   = w * 16;          // this wave's local column base
  const int n0   = nt * NT + cw;    // this wave's global column base

  const float  th = theta[t];
  const __bf16* Wt = WbfT + (size_t)t * K_ * M_;        // W^T[t], [k][m]
  const float*  Xb = X    + (size_t)b * M_ * N_;
  float*        Sb = S    + (size_t)b * K_ * N_;
  const __bf16* Ab = Abf  + (size_t)b * M_ * K_;        // bf16(A), [m][k]
  float*        Gb = gradA + (size_t)b * M_ * K_;

  // ---- phase 0: stage S tile into LDS (coalesced reads over n) ----
  for (int i = tid; i < NT * K_; i += WGT) {
    int n = i & (NT - 1);
    int k = i >> 7;
    ldsS[n * SP + k] = (__bf16)Sb[(size_t)k * N_ + nt * NT + n];
  }
  __syncthreads();

  // ---- phase 1: R = X - A*S  (per-wave private 16-column slice) ----
  for (int mt = 0; mt < 16; ++mt) {
    v8f acc = {};
#pragma unroll
    for (int kk = 0; kk < 8; ++kk) {
      v16bf af = frag_row(&Ab[(mt * 16 + nl) * K_ + kk * 32 + hi8]);   // A[m][k]
      v16bf bf = frag_row(&ldsS[(cw + nl) * SP + kk * 32 + hi8]);      // S[k][n]
      acc = wmma_bf16(af, bf, acc);
    }
    v8bf r;
#pragma unroll
    for (int rI = 0; rI < 8; ++rI) {
      int m = mt * 16 + hi8 + rI;
      r[rI] = (__bf16)(Xb[(size_t)m * N_ + n0 + nl] - acc[rI]);
    }
    *(v8bf*)&ldsR[(cw + nl) * SP + mt * 16 + hi8] = r;
  }

  // ---- phase 2: S = soft_threshold(S + W^T * R) ----
  for (int kt = 0; kt < 16; ++kt) {
    v8f acc = {};
#pragma unroll
    for (int mm = 0; mm < 8; ++mm) {
      v16bf af = frag_row(&Wt[(kt * 16 + nl) * M_ + mm * 32 + hi8]);   // W^T[k][m]
      v16bf bf = frag_row(&ldsR[(cw + nl) * SP + mm * 32 + hi8]);      // R[m][n]
      acc = wmma_bf16(af, bf, acc);
    }
    v8bf sv;
#pragma unroll
    for (int rI = 0; rI < 8; ++rI) {
      int k = kt * 16 + hi8 + rI;
      size_t idx = (size_t)k * N_ + n0 + nl;
      float s  = Sb[idx] + acc[rI];
      float mg = fabsf(s) - th;
      s = (mg > 0.0f) ? copysignf(mg, s) : 0.0f;
      Sb[idx] = s;                       // fp32 master (output)
      sv[rI]  = (__bf16)s;
      ldsSt[k * ST + cw + nl] = sv[rI];  // transposed copy for phase 4
    }
    *(v8bf*)&ldsS[(cw + nl) * SP + kt * 16 + hi8] = sv;
  }

  // ---- phase 3: E = A*S_new - X  (written n-contiguous for phase 4) ----
  for (int mt = 0; mt < 16; ++mt) {
    v8f acc = {};
#pragma unroll
    for (int kk = 0; kk < 8; ++kk) {
      v16bf af = frag_row(&Ab[(mt * 16 + nl) * K_ + kk * 32 + hi8]);
      v16bf bf = frag_row(&ldsS[(cw + nl) * SP + kk * 32 + hi8]);
      acc = wmma_bf16(af, bf, acc);
    }
#pragma unroll
    for (int rI = 0; rI < 8; ++rI) {
      int m = mt * 16 + hi8 + rI;
      ldsEt[m * ST + cw + nl] =
          (__bf16)(acc[rI] - Xb[(size_t)m * N_ + n0 + nl]);
    }
  }
  __syncthreads(); // phase 4 reads all columns of ldsEt / ldsSt

  // ---- phase 4: grad[m][k] += E[m][n] * S[k][n]^T  over this n-slab ----
#pragma unroll
  for (int ks = 0; ks < 2; ++ks) {
    int kt = w + ks * 8;               // each wave owns 2 k-tiles
    for (int mt = 0; mt < 16; ++mt) {
      v8f acc = {};
#pragma unroll
      for (int nn = 0; nn < 4; ++nn) {
        v16bf af = frag_row(&ldsEt[(mt * 16 + nl) * ST + nn * 32 + hi8]); // E: M=m, K=n
        v16bf bf = frag_row(&ldsSt[(kt * 16 + nl) * ST + nn * 32 + hi8]); // S^T: N=k, K=n
        acc = wmma_bf16(af, bf, acc);
      }
#pragma unroll
      for (int rI = 0; rI < 8; ++rI) {
        int m = mt * 16 + hi8 + rI;
        int k = kt * 16 + nl;
        unsafeAtomicAdd(&Gb[m * K_ + k], acc[rI]); // global_atomic_add_f32, L2-resident
      }
    }
  }
}

// ---------------------------------------------------------------------------
// K2: A = clip_cols(A - grad / L_A); refresh bf16(A); zero grad for next iter
// ---------------------------------------------------------------------------
__global__ __launch_bounds__(256, 1) void lpalm_updA(
    float* __restrict__ A, __bf16* __restrict__ Abf,
    float* __restrict__ gradA, const float* __restrict__ LA, int t)
{
  const int b = blockIdx.x;
  const int k = threadIdx.x;           // one column per thread; coalesced over k
  const float inv = 1.0f / LA[t];
  float*  Ab = A     + (size_t)b * M_ * K_;
  float*  Gb = gradA + (size_t)b * M_ * K_;
  __bf16* Fb = Abf   + (size_t)b * M_ * K_;

  float ss = 0.0f;
  for (int m = 0; m < M_; ++m) {
    int idx = m * K_ + k;
    float a = Ab[idx] - Gb[idx] * inv;
    Gb[idx] = 0.0f;
    Ab[idx] = a;
    ss += a * a;
  }
  float nrm = sqrtf(ss);
  float sc  = (nrm > 1.0f) ? (1.0f / nrm) : 1.0f;
  for (int m = 0; m < M_; ++m) {
    int idx = m * K_ + k;
    float a = Ab[idx] * sc;
    Ab[idx] = a;
    Fb[idx] = (__bf16)a;
  }
}

// ---------------------------------------------------------------------------
extern "C" void kernel_launch(void* const* d_in, const int* in_sizes, int n_in,
                              void* d_out, int out_size, void* d_ws, size_t ws_size,
                              hipStream_t stream)
{
  (void)in_sizes; (void)n_in; (void)out_size; (void)ws_size;
  const float* X       = (const float*)d_in[0]; // [B,M,N]
  const float* W       = (const float*)d_in[1]; // [T,M,K]
  const float* d_theta = (const float*)d_in[2]; // [T,1]
  const float* d_LA    = (const float*)d_in[3]; // [T,1]

  float* S = (float*)d_out;                     // output S_pred [B,K,N]
  float* A = S + (size_t)B_ * K_ * N_;          // output A_pred [B,M,K]

  // workspace: grad (4 MB) | bf16(A) (2 MB) | bf16(W^T) (1.25 MB)
  float*  gradA = (float*)d_ws;
  __bf16* Abf   = (__bf16*)((char*)d_ws + (size_t)B_ * M_ * K_ * 4);
  __bf16* WbfT  = (__bf16*)((char*)Abf  + (size_t)B_ * M_ * K_ * 2);

  lpalm_init<<<4096, 256, 0, stream>>>(S, A, Abf, gradA, WbfT, W);

  dim3 g1(N_ / NT, B_); // 16 n-tiles x 16 batches = 256 workgroups
  for (int t = 0; t < T_; ++t) {
    lpalm_iter<<<g1, WGT, 0, stream>>>(X, WbfT, d_theta, S, Abf, gradA, t);
    lpalm_updA<<<B_, 256, 0, stream>>>(A, Abf, gradA, d_LA, t);
  }
}
